// RNN_autoreg_18494129176876
// MI455X (gfx1250) — compile-verified, matrix operand order checked by reading, not run on hardware
//
#include <hip/hip_runtime.h>

// ---------------------------------------------------------------------------
// Problem constants (from the reference)
// ---------------------------------------------------------------------------
#define B_   4096
#define T_   60
#define NX_  4
#define NXS_ 3
#define NY_  4
#define NYS_ 3
#define H_   128
#define G4H_ 512            // 4*H gate width

// rnn1 fused K: [main(4) | mem(128) | zero-pad(28) | h(128)] = 288
#define K1C  288
#define LDA1 296            // LDS row stride (bf16 elems), 16B-aligned rows, conflict-friendly
// rnn2 fused K: [x(128) | h(128)] = 256
#define K2C  256
#define LDA2 264

typedef __attribute__((ext_vector_type(16))) __bf16 v16bf;
typedef __attribute__((ext_vector_type(8)))  float  v8f;
typedef __attribute__((ext_vector_type(4)))  __bf16 v4bf;

union Frag { v16bf v; float4 q[2]; };   // 32B: one WMMA A/B operand (16 bf16)
union P8   { float4 q; __bf16 b[8]; };

// Hardware tanh (gfx1250 V_TANH_F32) when the builtin exists; libm fallback.
__device__ __forceinline__ float htanh(float x) {
#if __has_builtin(__builtin_amdgcn_tanhf)
    return __builtin_amdgcn_tanhf(x);
#elif __has_builtin(__builtin_amdgcn_tanh_f32)
    return __builtin_amdgcn_tanh_f32(x);
#else
    return tanhf(x);
#endif
}
// sigmoid(x) = 0.5*tanh(x/2) + 0.5  -> single TRANS op + FMA
__device__ __forceinline__ float hsig(float x) { return fmaf(htanh(0.5f * x), 0.5f, 0.5f); }

// ---------------------------------------------------------------------------
// Kernel 0: pack weights once per launch.
//   Wc1 bf16[512][288] = [Wih1(132) | 0(28) | Whh1(128)]
//   Wc2 bf16[512][256] = [Wih2(128) | Whh2(128)]
//   bias1/2 = bih+bhh ; Wcomb = W_out @ W_lat ; bcomb = W_out@b_lat + b_out
// ---------------------------------------------------------------------------
__global__ void pack_kernel(const float* __restrict__ Wih1, const float* __restrict__ Whh1,
                            const float* __restrict__ bih1, const float* __restrict__ bhh1,
                            const float* __restrict__ Wih2, const float* __restrict__ Whh2,
                            const float* __restrict__ bih2, const float* __restrict__ bhh2,
                            const float* __restrict__ Wlat, const float* __restrict__ blat,
                            const float* __restrict__ Wout, const float* __restrict__ bout,
                            void* Wc1v, void* Wc2v, float* bias1, float* bias2,
                            float* Wcomb, float* bcomb)
{
    __bf16* Wc1 = (__bf16*)Wc1v;
    __bf16* Wc2 = (__bf16*)Wc2v;
    const int N1 = G4H_ * K1C, N2 = G4H_ * K2C;
    int idx = blockIdx.x * 256 + threadIdx.x;
    if (idx < N1) {
        int n = idx / K1C, k = idx % K1C;
        float v;
        if (k < 132)      v = Wih1[n * 132 + k];
        else if (k < 160) v = 0.0f;
        else              v = Whh1[n * H_ + (k - 160)];
        Wc1[idx] = (__bf16)v;
    } else if (idx < N1 + N2) {
        int t = idx - N1; int n = t / K2C, k = t % K2C;
        float v = (k < H_) ? Wih2[n * H_ + k] : Whh2[n * H_ + (k - H_)];
        Wc2[t] = (__bf16)v;
    } else if (idx < N1 + N2 + G4H_) {
        int n = idx - N1 - N2;
        bias1[n] = bih1[n] + bhh1[n];
    } else if (idx < N1 + N2 + 2 * G4H_) {
        int n = idx - N1 - N2 - G4H_;
        bias2[n] = bih2[n] + bhh2[n];
    } else if (idx < N1 + N2 + 2 * G4H_ + NY_ * H_) {
        int t = idx - (N1 + N2 + 2 * G4H_);
        int y = t >> 7, h = t & 127;
        float a = 0.0f;
        for (int j = 0; j < H_; ++j) a += Wout[y * H_ + j] * Wlat[j * H_ + h];
        Wcomb[t] = a;
    } else if (idx < N1 + N2 + 2 * G4H_ + NY_ * H_ + NY_) {
        int y = idx - (N1 + N2 + 2 * G4H_ + NY_ * H_);
        float a = bout[y];
        for (int j = 0; j < H_; ++j) a += Wout[y * H_ + j] * blat[j];
        bcomb[y] = a;
    }
}

// ---------------------------------------------------------------------------
// Kernel 1: rnn1 scan. One block = 32 batch rows, 8 waves.
// Wave w owns H-cols [16w,16w+16) for gates i,f,g,o; (h,c) stay in regs/LDS;
// weights get hoisted to persistent VGPRs by the compiler (loop-invariant).
// Next-step inputs are prefetched into registers during the WMMA phase.
// Writes h(step s) to r1out[:, T-1-s] (reference flips rnn1out back).
// ---------------------------------------------------------------------------
__global__ __launch_bounds__(256) void rnn1_kernel(
    const float* __restrict__ xmain, const float* __restrict__ sfc,
    const float* __restrict__ mem,
    const float* __restrict__ Wsf1, const float* __restrict__ bsf1,
    const float* __restrict__ Wsf2, const float* __restrict__ bsf2,
    const void* __restrict__ Wc1v, const float* __restrict__ bias1,
    void* __restrict__ r1outv)
{
    const __bf16* Wc1 = (const __bf16*)Wc1v;
    __bf16* r1out = (__bf16*)r1outv;

    __shared__ __bf16 As[32 * LDA1];
    const int tid = threadIdx.x;
    const int w = tid >> 5, lane = tid & 31, lh = lane & 15, hi = lane >> 4;
    const int b0 = blockIdx.x * 32;
    const int col = w * 16 + lh;                  // this lane's H column

    // zero the K pad (cols 132..159), written once
    for (int idx = tid; idx < 32 * 28; idx += 256)
        As[(idx / 28) * LDA1 + 132 + (idx % 28)] = (__bf16)0.0f;

    // initial h/c = tanh(sfc @ W_sfc{1,2}^T + b)
    float w1[3], w2[3];
    #pragma unroll
    for (int k = 0; k < 3; ++k) { w1[k] = Wsf1[col * 3 + k]; w2[k] = Wsf2[col * 3 + k]; }
    const float bb1 = bsf1[col], bb2 = bsf2[col];

    float cst[2][8], hr[2][8];
    #pragma unroll
    for (int m = 0; m < 2; ++m)
        #pragma unroll
        for (int r = 0; r < 8; ++r) {
            int row = b0 + m * 16 + r + hi * 8;
            float s0 = sfc[row * 3 + 0], s1 = sfc[row * 3 + 1], s2 = sfc[row * 3 + 2];
            hr[m][r]  = htanh(s0 * w1[0] + s1 * w1[1] + s2 * w1[2] + bb1);
            cst[m][r] = htanh(s0 * w2[0] + s1 * w2[1] + s2 * w2[2] + bb2);
        }

    // per-gate weight row pointers (loop-invariant -> compiler keeps frags in VGPRs)
    const __bf16* wrow[4];
    float bv[4];
    #pragma unroll
    for (int g = 0; g < 4; ++g) {
        wrow[g] = Wc1 + (size_t)(g * H_ + col) * K1C;
        bv[g]   = bias1[g * H_ + col];
    }

    // per-thread fill assignment
    const int frow = tid >> 3, fcb = (tid & 7) * 16;     // mem fill: row, col base
    const int mrow = tid >> 2, mcc = tid & 3;            // main fill (tid<128)

    // ---- prefetch step 0 inputs into registers ----
    float4 pm[4];
    float  xm = 0.0f;
    {
        const float* src = mem + ((size_t)(b0 + frow) * T_ + 0) * H_ + fcb;
        #pragma unroll
        for (int j = 0; j < 4; ++j) pm[j] = *(const float4*)(src + 4 * j);
        if (tid < 128)
            xm = xmain[((size_t)(b0 + mrow) * T_ + (T_ - 1)) * NX_ + mcc];
    }

    for (int s = 0; s < T_; ++s) {
        __syncthreads();                           // prev-step readers done
        // current h -> LDS A cols 160..287 (bf16)
        #pragma unroll
        for (int m = 0; m < 2; ++m)
            #pragma unroll
            for (int r = 0; r < 8; ++r)
                As[(m * 16 + r + hi * 8) * LDA1 + 160 + col] = (__bf16)hr[m][r];
        // flipped main input -> cols 0..3 (from prefetch regs)
        if (tid < 128)
            As[mrow * LDA1 + mcc] = (__bf16)xm;
        // rnn1_mem (not flipped) -> cols 4..131 (from prefetch regs)
        #pragma unroll
        for (int j = 0; j < 4; ++j) {
            v4bf t; t[0] = (__bf16)pm[j].x; t[1] = (__bf16)pm[j].y;
                    t[2] = (__bf16)pm[j].z; t[3] = (__bf16)pm[j].w;
            *(v4bf*)(&As[frow * LDA1 + 4 + fcb + 4 * j]) = t;
        }
        __syncthreads();

        // ---- issue next-step global loads; latency hidden behind WMMAs ----
        float4 pmN[4] = {pm[0], pm[1], pm[2], pm[3]};
        float  xmN = xm;
        if (s + 1 < T_) {
            const float* src = mem + ((size_t)(b0 + frow) * T_ + (s + 1)) * H_ + fcb;
            #pragma unroll
            for (int j = 0; j < 4; ++j) pmN[j] = *(const float4*)(src + 4 * j);
            if (tid < 128)
                xmN = xmain[((size_t)(b0 + mrow) * T_ + (T_ - 2 - s)) * NX_ + mcc];
        }

        v8f acc[2][4];
        v8f z = {0.f, 0.f, 0.f, 0.f, 0.f, 0.f, 0.f, 0.f};
        #pragma unroll
        for (int m = 0; m < 2; ++m)
            #pragma unroll
            for (int g = 0; g < 4; ++g) acc[m][g] = z;

        #pragma unroll
        for (int kc = 0; kc < K1C / 32; ++kc) {
            const int k0 = kc * 32 + hi * 8;
            Frag a0, a1;
            a0.q[0] = *(const float4*)(&As[lh * LDA1 + k0]);
            a0.q[1] = *(const float4*)(&As[lh * LDA1 + k0 + 16]);
            a1.q[0] = *(const float4*)(&As[(16 + lh) * LDA1 + k0]);
            a1.q[1] = *(const float4*)(&As[(16 + lh) * LDA1 + k0 + 16]);
            #pragma unroll
            for (int g = 0; g < 4; ++g) {
                Frag b;
                b.q[0] = *(const float4*)(wrow[g] + k0);
                b.q[1] = *(const float4*)(wrow[g] + k0 + 16);
                acc[0][g] = __builtin_amdgcn_wmma_f32_16x16x32_bf16(
                    false, a0.v, false, b.v, (short)0, acc[0][g], false, false);
                acc[1][g] = __builtin_amdgcn_wmma_f32_16x16x32_bf16(
                    false, a1.v, false, b.v, (short)0, acc[1][g], false, false);
            }
        }

        // pointwise LSTM update (gate order i,f,g,o), store h at flipped time
        #pragma unroll
        for (int m = 0; m < 2; ++m)
            #pragma unroll
            for (int r = 0; r < 8; ++r) {
                float iv = acc[m][0][r] + bv[0];
                float fv = acc[m][1][r] + bv[1];
                float gv = acc[m][2][r] + bv[2];
                float ov = acc[m][3][r] + bv[3];
                float cn = hsig(fv) * cst[m][r] + hsig(iv) * htanh(gv);
                float hn = hsig(ov) * htanh(cn);
                cst[m][r] = cn; hr[m][r] = hn;
                int row = b0 + m * 16 + r + hi * 8;
                r1out[((size_t)row * T_ + (T_ - 1 - s)) * H_ + col] = (__bf16)hn;
            }

        #pragma unroll
        for (int j = 0; j < 4; ++j) pm[j] = pmN[j];
        xm = xmN;
    }
}

// ---------------------------------------------------------------------------
// Kernel 2: rnn2 scan over rnn1out (time-ordered), init from hx2/cx2.
// ---------------------------------------------------------------------------
__global__ __launch_bounds__(256) void rnn2_kernel(
    const void* __restrict__ r1outv,
    const float* __restrict__ hx2, const float* __restrict__ cx2,
    const void* __restrict__ Wc2v, const float* __restrict__ bias2,
    void* __restrict__ r2outv)
{
    const __bf16* r1out = (const __bf16*)r1outv;
    const __bf16* Wc2 = (const __bf16*)Wc2v;
    __bf16* r2out = (__bf16*)r2outv;

    __shared__ __bf16 As[32 * LDA2];
    const int tid = threadIdx.x;
    const int w = tid >> 5, lane = tid & 31, lh = lane & 15, hi = lane >> 4;
    const int b0 = blockIdx.x * 32;
    const int col = w * 16 + lh;

    float cst[2][8], hr[2][8];
    #pragma unroll
    for (int m = 0; m < 2; ++m)
        #pragma unroll
        for (int r = 0; r < 8; ++r) {
            int row = b0 + m * 16 + r + hi * 8;
            hr[m][r]  = hx2[(size_t)row * H_ + col];
            cst[m][r] = cx2[(size_t)row * H_ + col];
        }

    const __bf16* wrow[4];
    float bv[4];
    #pragma unroll
    for (int g = 0; g < 4; ++g) {
        wrow[g] = Wc2 + (size_t)(g * H_ + col) * K2C;
        bv[g]   = bias2[g * H_ + col];
    }

    const int frow = tid >> 3, fcb = (tid & 7) * 16;

    // prefetch step 0 x-row (raw bf16, 32B/thread)
    float4 px[2];
    {
        const __bf16* src = r1out + ((size_t)(b0 + frow) * T_ + 0) * H_ + fcb;
        px[0] = *(const float4*)(src);
        px[1] = *(const float4*)(src + 8);
    }

    for (int s = 0; s < T_; ++s) {
        __syncthreads();
        #pragma unroll
        for (int m = 0; m < 2; ++m)
            #pragma unroll
            for (int r = 0; r < 8; ++r)
                As[(m * 16 + r + hi * 8) * LDA2 + 128 + col] = (__bf16)hr[m][r];
        // x tile from prefetch regs -> LDS cols 0..127
        *(float4*)(&As[frow * LDA2 + fcb])     = px[0];
        *(float4*)(&As[frow * LDA2 + fcb + 8]) = px[1];
        __syncthreads();

        float4 pxN[2] = {px[0], px[1]};
        if (s + 1 < T_) {
            const __bf16* src = r1out + ((size_t)(b0 + frow) * T_ + (s + 1)) * H_ + fcb;
            pxN[0] = *(const float4*)(src);
            pxN[1] = *(const float4*)(src + 8);
        }

        v8f acc[2][4];
        v8f z = {0.f, 0.f, 0.f, 0.f, 0.f, 0.f, 0.f, 0.f};
        #pragma unroll
        for (int m = 0; m < 2; ++m)
            #pragma unroll
            for (int g = 0; g < 4; ++g) acc[m][g] = z;

        #pragma unroll
        for (int kc = 0; kc < K2C / 32; ++kc) {
            const int k0 = kc * 32 + hi * 8;
            Frag a0, a1;
            a0.q[0] = *(const float4*)(&As[lh * LDA2 + k0]);
            a0.q[1] = *(const float4*)(&As[lh * LDA2 + k0 + 16]);
            a1.q[0] = *(const float4*)(&As[(16 + lh) * LDA2 + k0]);
            a1.q[1] = *(const float4*)(&As[(16 + lh) * LDA2 + k0 + 16]);
            #pragma unroll
            for (int g = 0; g < 4; ++g) {
                Frag b;
                b.q[0] = *(const float4*)(wrow[g] + k0);
                b.q[1] = *(const float4*)(wrow[g] + k0 + 16);
                acc[0][g] = __builtin_amdgcn_wmma_f32_16x16x32_bf16(
                    false, a0.v, false, b.v, (short)0, acc[0][g], false, false);
                acc[1][g] = __builtin_amdgcn_wmma_f32_16x16x32_bf16(
                    false, a1.v, false, b.v, (short)0, acc[1][g], false, false);
            }
        }

        #pragma unroll
        for (int m = 0; m < 2; ++m)
            #pragma unroll
            for (int r = 0; r < 8; ++r) {
                float iv = acc[m][0][r] + bv[0];
                float fv = acc[m][1][r] + bv[1];
                float gv = acc[m][2][r] + bv[2];
                float ov = acc[m][3][r] + bv[3];
                float cn = hsig(fv) * cst[m][r] + hsig(iv) * htanh(gv);
                float hn = hsig(ov) * htanh(cn);
                cst[m][r] = cn; hr[m][r] = hn;
                int row = b0 + m * 16 + r + hi * 8;
                r2out[((size_t)row * T_ + s) * H_ + col] = (__bf16)hn;
            }

        px[0] = pxN[0]; px[1] = pxN[1];
    }
}

// ---------------------------------------------------------------------------
// Kernel 3: out = rnn2out @ (W_out@W_lat)^T + bcomb.
// One thread per (b,t): row read once, NY=4 dots, one float4 store.
// ---------------------------------------------------------------------------
__global__ void out_kernel(const void* __restrict__ r2outv,
                           const float* __restrict__ Wcomb,
                           const float* __restrict__ bcomb,
                           float* __restrict__ out)
{
    const __bf16* r2out = (const __bf16*)r2outv;
    int bt = blockIdx.x * 256 + threadIdx.x;
    if (bt >= B_ * T_) return;
    const __bf16* hrow = r2out + (size_t)bt * H_;
    float a[NY_] = {0.f, 0.f, 0.f, 0.f};
    #pragma unroll 4
    for (int k0 = 0; k0 < H_; k0 += 8) {
        P8 p; p.q = *(const float4*)(hrow + k0);
        #pragma unroll
        for (int j = 0; j < 8; ++j) {
            float x = (float)p.b[j];
            #pragma unroll
            for (int y = 0; y < NY_; ++y) a[y] += x * Wcomb[y * H_ + k0 + j];
        }
    }
    float4 o;
    o.x = a[0] + bcomb[0]; o.y = a[1] + bcomb[1];
    o.z = a[2] + bcomb[2]; o.w = a[3] + bcomb[3];
    *(float4*)(out + (size_t)bt * NY_) = o;
}

// ---------------------------------------------------------------------------
// Kernel 4: out_sfc = last_h @ W_so^T + b_so  (last_h = rnn2out[:, T-1])
// ---------------------------------------------------------------------------
__global__ void sfc_out_kernel(const void* __restrict__ r2outv,
                               const float* __restrict__ Wso,
                               const float* __restrict__ bso,
                               float* __restrict__ out)
{
    const __bf16* r2out = (const __bf16*)r2outv;
    int idx = blockIdx.x * 256 + threadIdx.x;
    if (idx >= B_ * NYS_) return;
    int b = idx / NYS_, j = idx - b * NYS_;
    const __bf16* hrow = r2out + ((size_t)b * T_ + (T_ - 1)) * H_;
    const float* wr = Wso + j * H_;
    float a = 0.0f;
    #pragma unroll
    for (int k0 = 0; k0 < H_; k0 += 8) {
        P8 p; p.q = *(const float4*)(hrow + k0);
        #pragma unroll
        for (int jj = 0; jj < 8; ++jj) a += (float)p.b[jj] * wr[k0 + jj];
    }
    out[(size_t)B_ * T_ * NY_ + idx] = a + bso[j];
}

// ---------------------------------------------------------------------------
// Workspace layout (bytes): ~126.4 MB total
// ---------------------------------------------------------------------------
static const size_t WC1_OFF   = 0;                         // 512*288*2 = 294912
static const size_t WC2_OFF   = 294912;                    // 512*256*2 = 262144
static const size_t B1_OFF    = 557056;                    // 512 f32
static const size_t B2_OFF    = 559104;                    // 512 f32
static const size_t WCOMB_OFF = 561152;                    // 4*128 f32
static const size_t BCOMB_OFF = 563200;                    // 4 f32
static const size_t R1_OFF    = 563456;                    // B*T*H bf16 = 62914560
static const size_t R2_OFF    = 563456 + 62914560ull;      // B*T*H bf16

extern "C" void kernel_launch(void* const* d_in, const int* in_sizes, int n_in,
                              void* d_out, int out_size, void* d_ws, size_t ws_size,
                              hipStream_t stream)
{
    (void)in_sizes; (void)n_in; (void)out_size; (void)ws_size;
    const float* xmain = (const float*)d_in[0];
    const float* sfc   = (const float*)d_in[1];
    const float* mem   = (const float*)d_in[2];
    const float* hx2   = (const float*)d_in[3];
    const float* cx2   = (const float*)d_in[4];
    const float* Wsf1  = (const float*)d_in[5];
    const float* bsf1  = (const float*)d_in[6];
    const float* Wsf2  = (const float*)d_in[7];
    const float* bsf2  = (const float*)d_in[8];
    const float* Wih1  = (const float*)d_in[9];
    const float* Whh1  = (const float*)d_in[10];
    const float* bih1  = (const float*)d_in[11];
    const float* bhh1  = (const float*)d_in[12];
    const float* Wih2  = (const float*)d_in[13];
    const float* Whh2  = (const float*)d_in[14];
    const float* bih2  = (const float*)d_in[15];
    const float* bhh2  = (const float*)d_in[16];
    const float* Wlat  = (const float*)d_in[17];
    const float* blat  = (const float*)d_in[18];
    const float* Wout  = (const float*)d_in[19];
    const float* bout  = (const float*)d_in[20];
    const float* Wso   = (const float*)d_in[21];
    const float* bso   = (const float*)d_in[22];

    char* ws = (char*)d_ws;
    void*  Wc1   = (void*)(ws + WC1_OFF);
    void*  Wc2   = (void*)(ws + WC2_OFF);
    float* bias1 = (float*)(ws + B1_OFF);
    float* bias2 = (float*)(ws + B2_OFF);
    float* Wcomb = (float*)(ws + WCOMB_OFF);
    float* bcomb = (float*)(ws + BCOMB_OFF);
    void*  r1out = (void*)(ws + R1_OFF);
    void*  r2out = (void*)(ws + R2_OFF);

    const int packN = G4H_ * K1C + G4H_ * K2C + 2 * G4H_ + NY_ * H_ + NY_;
    pack_kernel<<<(packN + 255) / 256, 256, 0, stream>>>(
        Wih1, Whh1, bih1, bhh1, Wih2, Whh2, bih2, bhh2,
        Wlat, blat, Wout, bout, Wc1, Wc2, bias1, bias2, Wcomb, bcomb);

    rnn1_kernel<<<B_ / 32, 256, 0, stream>>>(
        xmain, sfc, mem, Wsf1, bsf1, Wsf2, bsf2, Wc1, bias1, r1out);

    rnn2_kernel<<<B_ / 32, 256, 0, stream>>>(
        r1out, hx2, cx2, Wc2, bias2, r2out);

    out_kernel<<<(B_ * T_ + 255) / 256, 256, 0, stream>>>(
        r2out, Wcomb, bcomb, (float*)d_out);

    sfc_out_kernel<<<(B_ * NYS_ + 255) / 256, 256, 0, stream>>>(
        r2out, Wso, bso, (float*)d_out);
}
